// Decoder_65326452572923
// MI455X (gfx1250) — matrix operand order repaired and checked by reading, not compiled, not run on hardware
//
#include <hip/hip_runtime.h>
#include <stdint.h>

#define N_NODES 41616
#define N_EDGES 249696
#define NBLK 18

typedef __attribute__((ext_vector_type(16))) __bf16 bf16x16;
typedef __attribute__((ext_vector_type(8)))  float  v8f;

union ABReg { bf16x16 v; uint4 u[2]; };

static __device__ __forceinline__ unsigned int f2bf(float f) {
  unsigned int u = __float_as_uint(f);
  return (u + 0x7FFFu + ((u >> 16) & 1u)) >> 16;   // round-to-nearest-even
}

#define CDIV(a, b) (((a) + (b) - 1) / (b))

// ---------------- degree / edge weights ----------------
__global__ void deg_count(const int* __restrict__ dst, float* __restrict__ deg) {
  int e = blockIdx.x * blockDim.x + threadIdx.x;
  if (e < N_EDGES) atomicAdd(&deg[dst[e]], 1.0f);
}
__global__ void dinv_k(const float* __restrict__ deg, float* __restrict__ dinv) {
  int i = blockIdx.x * blockDim.x + threadIdx.x;
  if (i < N_NODES) {
    float d = deg[i];
    dinv[i] = d > 0.0f ? rsqrtf(fmaxf(d, 1.0f)) : 0.0f;
  }
}
__global__ void edge_w(const int* __restrict__ src, const int* __restrict__ dst,
                       const float* __restrict__ dinv, float* __restrict__ w) {
  int e = blockIdx.x * blockDim.x + threadIdx.x;
  if (e < N_EDGES) w[e] = -dinv[src[e]] * dinv[dst[e]];
}

// ---------------- MLP ----------------
__global__ void mlp1(const float* __restrict__ vec, const float* __restrict__ W,
                     const float* __restrict__ b, float* __restrict__ h1) {
  int t = blockIdx.x * blockDim.x + threadIdx.x;
  if (t >= 16 * 128) return;
  int g = t >> 7, c = t & 127;
  float acc = b[c];
  #pragma unroll
  for (int i = 0; i < 32; ++i) acc += vec[g * 32 + i] * W[i * 128 + c];
  h1[t] = fmaxf(acc, 0.0f);
}
__global__ void mlp2(const float* __restrict__ h1, const float* __restrict__ W,
                     const float* __restrict__ b, float* __restrict__ h2) {
  int t = blockIdx.x * blockDim.x + threadIdx.x;
  if (t >= 16 * 15606) return;
  int g = t / 15606, p = t - g * 15606;
  float acc = b[p];
  #pragma unroll 8
  for (int i = 0; i < 128; ++i) acc += h1[g * 128 + i] * W[(size_t)i * 15606 + p];
  h2[t] = fmaxf(acc, 0.0f);
}
__global__ void build_x9(const float* __restrict__ sp, const float* __restrict__ h2,
                         float* __restrict__ x9) {
  int t = blockIdx.x * blockDim.x + threadIdx.x;
  if (t >= N_NODES * 16) return;
  int n = t >> 4, c = t & 15;
  if (c >= 9) return;
  float v;
  if (c < 3) v = sp[n * 3 + c];
  else {
    int g = n / 2601, p = n - g * 2601;
    v = h2[(size_t)g * 15606 + p * 6 + (c - 3)];
  }
  x9[(size_t)n * 9 + c] = v;
}

// ---------------- 9-channel propagation (initial conv) ----------------
__global__ void scatter9(float* __restrict__ y, const float* __restrict__ x,
                         const int* __restrict__ src, const int* __restrict__ dst,
                         const float* __restrict__ w) {
  int t = blockIdx.x * blockDim.x + threadIdx.x;
  if (t >= N_EDGES * 16) return;
  int e = t >> 4, c = t & 15;
  if (c >= 9) return;
  atomicAdd(&y[(size_t)dst[e] * 9 + c], w[e] * x[(size_t)src[e] * 9 + c]);
}
__global__ void combine9(float* __restrict__ p, const float* __restrict__ told) {
  int t = blockIdx.x * blockDim.x + threadIdx.x;
  if (t < N_NODES * 9) p[t] = 2.0f * p[t] - told[t];
}
__global__ void copy9slot(const float* __restrict__ src, float* __restrict__ T9, int slot) {
  int t = blockIdx.x * blockDim.x + threadIdx.x;
  if (t >= N_NODES * 9) return;
  int n = t / 9, c = t - n * 9;
  T9[(size_t)n * 54 + slot * 9 + c] = src[t];
}
__global__ __launch_bounds__(128)
void cheb_init_gemm(const float* __restrict__ T9, const float* __restrict__ Wi,
                    const float* __restrict__ bi, float* __restrict__ out) {
  __shared__ float row[54];
  int n = blockIdx.x, c = threadIdx.x;
  if (c < 54) row[c] = T9[(size_t)n * 54 + c];
  __syncthreads();
  float acc = bi[c];
  #pragma unroll
  for (int j = 0; j < 54; ++j) acc += row[j] * Wi[j * 128 + c];
  out[(size_t)n * 128 + c] = acc;
}

// ---------------- 128-channel propagation ----------------
__global__ void scatter128(float* __restrict__ y, const float* __restrict__ x,
                           const int* __restrict__ src, const int* __restrict__ dst,
                           const float* __restrict__ w) {
  int t = blockIdx.x * blockDim.x + threadIdx.x;
  if (t >= N_EDGES * 32) return;
  int e = t >> 5;
  int c4 = (t & 31) * 4;
  float we = w[e];
  float4 xv = *(const float4*)(x + (size_t)src[e] * 128 + c4);
  float* yd = y + (size_t)dst[e] * 128 + c4;
  atomicAdd(yd + 0, we * xv.x);
  atomicAdd(yd + 1, we * xv.y);
  atomicAdd(yd + 2, we * xv.z);
  atomicAdd(yd + 3, we * xv.w);
}
__global__ void combine4(float* __restrict__ p, const float* __restrict__ told) {
  int i = blockIdx.x * blockDim.x + threadIdx.x;
  if (i >= N_NODES * 32) return;
  float4 a = ((float4*)p)[i];
  float4 b = ((const float4*)told)[i];
  a.x = 2.0f * a.x - b.x; a.y = 2.0f * a.y - b.y;
  a.z = 2.0f * a.z - b.z; a.w = 2.0f * a.w - b.w;
  ((float4*)p)[i] = a;
}

// f32 [N][128] -> bf16 Tall[N][768] at kcheb slot
__global__ void cvt_slot(const float* __restrict__ src, uint32_t* __restrict__ tall, int slot) {
  int t = blockIdx.x * blockDim.x + threadIdx.x;
  if (t >= N_NODES * 64) return;
  int i = t * 2;
  int n = i >> 7, c = i & 127;
  float2 f = *(const float2*)(src + (size_t)n * 128 + c);
  tall[((size_t)n * 768 + slot * 128 + c) >> 1] = f2bf(f.x) | (f2bf(f.y) << 16);
}

// pack Wb (f32 [18][6][128][128]) into WMMA B-operand register image (bf16)
__global__ void packW(const float* __restrict__ Wb, uint32_t* __restrict__ Wp) {
  int t = blockIdx.x * blockDim.x + threadIdx.x;
  if (t >= NBLK * 6 * 8192) return;
  int r = t;
  int v  = r & 7;  r >>= 3;
  int l  = r & 31; r >>= 5;
  int nt = r & 7;  r >>= 3;
  int ks = r & 3;  r >>= 2;
  int kc = r % 6;
  int cc = r / 6;
  int n   = nt * 16 + (l & 15);
  int khi = l >> 4;
  int k0  = ks * 32 + khi * 16 + 2 * v;
  size_t base = ((size_t)cc * 6 + kc) * 128;
  float w0 = Wb[(base + k0) * 128 + n];
  float w1 = Wb[(base + k0 + 1) * 128 + n];
  Wp[t] = f2bf(w0) | (f2bf(w1) << 16);
}

// ---------------- batched Chebyshev GEMM: out = [Tall|K=768] @ Wpack + bias ----------------
__global__ __launch_bounds__(256)
void gemm128(const unsigned short* __restrict__ Tall, const uint32_t* __restrict__ Wp,
             const float* __restrict__ bias, const float* __restrict__ res,
             float* __restrict__ out, int relu) {
  const int lane  = threadIdx.x & 31;
  const int ntile = threadIdx.x >> 5;      // 8 waves = 8 column tiles
  const int mtile = blockIdx.x;            // 2601 row tiles of 16
  const int lrow  = lane & 15;
  const int khi   = lane >> 4;
  const unsigned short* arow = Tall + (size_t)(mtile * 16 + lrow) * 768;
  v8f acc = {0.f, 0.f, 0.f, 0.f, 0.f, 0.f, 0.f, 0.f};
  #pragma unroll
  for (int kc = 0; kc < 6; ++kc) {
    #pragma unroll
    for (int ks = 0; ks < 4; ++ks) {
      ABReg a, b;
      int offA = kc * 128 + ks * 32 + 16 * khi;
      a.u[0] = *(const uint4*)(arow + offA);
      a.u[1] = *(const uint4*)(arow + offA + 8);
      const uint32_t* bp = Wp + ((((kc * 4 + ks) * 8 + ntile) * 32 + lane) * 8);
      b.u[0] = *(const uint4*)(bp);
      b.u[1] = *(const uint4*)(bp + 4);
      acc = __builtin_amdgcn_wmma_f32_16x16x32_bf16(false, a.v, false, b.v,
                                                    (short)0, acc, false, false);
    }
  }
  const int col = ntile * 16 + lrow;
  const float bc = bias[col];
  #pragma unroll
  for (int v = 0; v < 8; ++v) {
    int row = mtile * 16 + v + 8 * khi;
    float val = acc[v] + bc;
    if (relu) val = fmaxf(val, 0.0f);
    if (res) val += res[(size_t)row * 128 + col];
    out[(size_t)row * 128 + col] = val;
  }
}

// ---------------- final conv (128 -> 3) + tanh ----------------
__global__ void final_acc(const float* __restrict__ Tk, const float* __restrict__ Wf,
                          int kk, float* __restrict__ out3) {
  int t = blockIdx.x * blockDim.x + threadIdx.x;
  if (t >= N_NODES * 3) return;
  int n = t / 3, c = t - n * 3;
  const float* xr = Tk + (size_t)n * 128;
  float acc = 0.0f;
  #pragma unroll 8
  for (int f = 0; f < 128; ++f) acc += xr[f] * Wf[(kk * 128 + f) * 3 + c];
  out3[t] += acc;
}
__global__ void tanh_out(const float* __restrict__ out3, const float* __restrict__ bf,
                         float* __restrict__ out) {
  int t = blockIdx.x * blockDim.x + threadIdx.x;
  if (t >= N_NODES * 3) return;
  out[t] = tanhf(out3[t] + bf[t % 3]);
}

// ---------------- host orchestration ----------------
extern "C" void kernel_launch(void* const* d_in, const int* in_sizes, int n_in,
                              void* d_out, int out_size, void* d_ws, size_t ws_size,
                              hipStream_t stream) {
  const float* vector = (const float*)d_in[0];
  const int*   edges  = (const int*)d_in[1];
  const float* sphere = (const float*)d_in[2];
  const float* fc1_w  = (const float*)d_in[3];
  const float* fc1_b  = (const float*)d_in[4];
  const float* fc2_w  = (const float*)d_in[5];
  const float* fc2_b  = (const float*)d_in[6];
  const float* Wi     = (const float*)d_in[7];
  const float* bi     = (const float*)d_in[8];
  const float* Wb     = (const float*)d_in[9];
  const float* bb     = (const float*)d_in[10];
  const float* Wf     = (const float*)d_in[11];
  const float* bfv    = (const float*)d_in[12];
  const int* src = edges;
  const int* dst = edges + N_EDGES;

  char* ws = (char*)d_ws;
  size_t off = 0;
  auto alloc = [&](size_t bytes) -> char* {
    char* p = ws + off;
    off = (off + bytes + 255) & ~(size_t)255;
    return p;
  };
  const size_t nb128 = (size_t)N_NODES * 128 * 4;
  const size_t nb9   = (size_t)N_NODES * 9 * 4;

  float* deg   = (float*)alloc((size_t)N_NODES * 4);
  float* dinv  = (float*)alloc((size_t)N_NODES * 4);
  float* w     = (float*)alloc((size_t)N_EDGES * 4);
  float* h1    = (float*)alloc(16 * 128 * 4);
  float* h2    = (float*)alloc((size_t)16 * 15606 * 4);
  float* x9    = (float*)alloc(nb9);
  float* t9a   = (float*)alloc(nb9);
  float* t9b   = (float*)alloc(nb9);
  float* t9c   = (float*)alloc(nb9);
  float* T9    = (float*)alloc((size_t)N_NODES * 54 * 4);
  float* Xa    = (float*)alloc(nb128);
  float* Xb    = (float*)alloc(nb128);
  float* B1    = (float*)alloc(nb128);
  float* B2    = (float*)alloc(nb128);
  float* B3    = (float*)alloc(nb128);
  uint32_t* Tall = (uint32_t*)alloc((size_t)N_NODES * 768 * 2);
  uint32_t* Wp   = (uint32_t*)alloc((size_t)NBLK * 6 * 8192 * 4);
  float* out3  = (float*)alloc((size_t)N_NODES * 3 * 4);

  const int TB = 256;
  const int grE   = CDIV(N_EDGES, TB);
  const int grN   = CDIV(N_NODES, TB);
  const int grE32 = CDIV(N_EDGES * 32, TB);
  const int grE16 = CDIV(N_EDGES * 16, TB);
  const int grN9  = CDIV(N_NODES * 9, TB);
  const int grN16 = CDIV(N_NODES * 16, TB);
  const int grN32 = CDIV(N_NODES * 32, TB);
  const int grN64 = CDIV(N_NODES * 64, TB);
  const int grN3  = CDIV(N_NODES * 3, TB);

  // edge weights
  hipMemsetAsync(deg, 0, (size_t)N_NODES * 4, stream);
  deg_count<<<grE, TB, 0, stream>>>(dst, deg);
  dinv_k<<<grN, TB, 0, stream>>>(deg, dinv);
  edge_w<<<grE, TB, 0, stream>>>(src, dst, dinv, w);

  // MLP -> x9
  mlp1<<<CDIV(16 * 128, TB), TB, 0, stream>>>(vector, fc1_w, fc1_b, h1);
  mlp2<<<CDIV(16 * 15606, TB), TB, 0, stream>>>(h1, fc2_w, fc2_b, h2);
  build_x9<<<grN16, TB, 0, stream>>>(sphere, h2, x9);

  // pack all block weights for WMMA B operands
  packW<<<CDIV(NBLK * 6 * 8192, TB), TB, 0, stream>>>(Wb, Wp);

  // initial ChebConv (9 -> 128)
  copy9slot<<<grN9, TB, 0, stream>>>(x9, T9, 0);
  hipMemsetAsync(t9b, 0, nb9, stream);
  scatter9<<<grE16, TB, 0, stream>>>(t9b, x9, src, dst, w);
  copy9slot<<<grN9, TB, 0, stream>>>(t9b, T9, 1);
  hipMemsetAsync(t9c, 0, nb9, stream);
  scatter9<<<grE16, TB, 0, stream>>>(t9c, t9b, src, dst, w);
  combine9<<<grN9, TB, 0, stream>>>(t9c, x9);
  copy9slot<<<grN9, TB, 0, stream>>>(t9c, T9, 2);
  hipMemsetAsync(t9a, 0, nb9, stream);
  scatter9<<<grE16, TB, 0, stream>>>(t9a, t9c, src, dst, w);
  combine9<<<grN9, TB, 0, stream>>>(t9a, t9b);
  copy9slot<<<grN9, TB, 0, stream>>>(t9a, T9, 3);
  hipMemsetAsync(t9b, 0, nb9, stream);
  scatter9<<<grE16, TB, 0, stream>>>(t9b, t9a, src, dst, w);
  combine9<<<grN9, TB, 0, stream>>>(t9b, t9c);
  copy9slot<<<grN9, TB, 0, stream>>>(t9b, T9, 4);
  hipMemsetAsync(t9c, 0, nb9, stream);
  scatter9<<<grE16, TB, 0, stream>>>(t9c, t9b, src, dst, w);
  combine9<<<grN9, TB, 0, stream>>>(t9c, t9a);
  copy9slot<<<grN9, TB, 0, stream>>>(t9c, T9, 5);
  cheb_init_gemm<<<N_NODES, 128, 0, stream>>>(T9, Wi, bi, Xa);

  // one 128->128 ChebConv: stage 6 Chebyshev terms as bf16, single batched WMMA GEMM
  auto conv128 = [&](const float* xin, float* xout, int cc, const float* res, int relu) {
    cvt_slot<<<grN64, TB, 0, stream>>>(xin, Tall, 0);
    hipMemsetAsync(B1, 0, nb128, stream);
    scatter128<<<grE32, TB, 0, stream>>>(B1, xin, src, dst, w);
    cvt_slot<<<grN64, TB, 0, stream>>>(B1, Tall, 1);
    hipMemsetAsync(B2, 0, nb128, stream);
    scatter128<<<grE32, TB, 0, stream>>>(B2, B1, src, dst, w);
    combine4<<<grN32, TB, 0, stream>>>(B2, xin);
    cvt_slot<<<grN64, TB, 0, stream>>>(B2, Tall, 2);
    hipMemsetAsync(B3, 0, nb128, stream);
    scatter128<<<grE32, TB, 0, stream>>>(B3, B2, src, dst, w);
    combine4<<<grN32, TB, 0, stream>>>(B3, B1);
    cvt_slot<<<grN64, TB, 0, stream>>>(B3, Tall, 3);
    hipMemsetAsync(B1, 0, nb128, stream);
    scatter128<<<grE32, TB, 0, stream>>>(B1, B3, src, dst, w);
    combine4<<<grN32, TB, 0, stream>>>(B1, B2);
    cvt_slot<<<grN64, TB, 0, stream>>>(B1, Tall, 4);
    hipMemsetAsync(B2, 0, nb128, stream);
    scatter128<<<grE32, TB, 0, stream>>>(B2, B1, src, dst, w);
    combine4<<<grN32, TB, 0, stream>>>(B2, B3);
    cvt_slot<<<grN64, TB, 0, stream>>>(B2, Tall, 5);
    gemm128<<<N_NODES / 16, 256, 0, stream>>>((const unsigned short*)Tall,
                                              Wp + (size_t)cc * 49152,
                                              bb + (size_t)cc * 128, res, xout, relu);
  };

  // 6 residual blocks of 3 convs
  float* cur = Xa;
  float* alt = Xb;
  for (int i = 0; i < 6; ++i) {
    conv128(cur, alt, 3 * i + 0, nullptr, 1);
    conv128(alt, alt, 3 * i + 1, nullptr, 1);
    conv128(alt, alt, 3 * i + 2, cur, 1);   // relu then + residual
    float* tmp = cur; cur = alt; alt = tmp;
  }

  // final ChebConv (128 -> 3) + tanh
  hipMemsetAsync(out3, 0, (size_t)N_NODES * 3 * 4, stream);
  final_acc<<<grN3, TB, 0, stream>>>(cur, Wf, 0, out3);
  hipMemsetAsync(B1, 0, nb128, stream);
  scatter128<<<grE32, TB, 0, stream>>>(B1, cur, src, dst, w);
  final_acc<<<grN3, TB, 0, stream>>>(B1, Wf, 1, out3);
  hipMemsetAsync(B2, 0, nb128, stream);
  scatter128<<<grE32, TB, 0, stream>>>(B2, B1, src, dst, w);
  combine4<<<grN32, TB, 0, stream>>>(B2, cur);
  final_acc<<<grN3, TB, 0, stream>>>(B2, Wf, 2, out3);
  hipMemsetAsync(B3, 0, nb128, stream);
  scatter128<<<grE32, TB, 0, stream>>>(B3, B2, src, dst, w);
  combine4<<<grN32, TB, 0, stream>>>(B3, B1);
  final_acc<<<grN3, TB, 0, stream>>>(B3, Wf, 3, out3);
  hipMemsetAsync(B1, 0, nb128, stream);
  scatter128<<<grE32, TB, 0, stream>>>(B1, B3, src, dst, w);
  combine4<<<grN32, TB, 0, stream>>>(B1, B2);
  final_acc<<<grN3, TB, 0, stream>>>(B1, Wf, 4, out3);
  hipMemsetAsync(B2, 0, nb128, stream);
  scatter128<<<grE32, TB, 0, stream>>>(B2, B1, src, dst, w);
  combine4<<<grN32, TB, 0, stream>>>(B2, B3);
  final_acc<<<grN3, TB, 0, stream>>>(B2, Wf, 5, out3);

  tanh_out<<<grN3, TB, 0, stream>>>(out3, bfv, (float*)d_out);

  (void)in_sizes; (void)n_in; (void)out_size; (void)ws_size;
}
